// UnifiedQLSTMCell_65481071404611
// MI455X (gfx1250) — compile-verified, matrix operand order checked by reading, not run on hardware
//
#include <hip/hip_runtime.h>
#include <hip/hip_bf16.h>

typedef __attribute__((ext_vector_type(16))) _Float16 v16h;
typedef __attribute__((ext_vector_type(8)))  float    v8f;

#define BATCH   8192
#define IN_DIM  128
#define NQ      10
#define KDIM    138   // 128 + 10
#define KPAD    160   // 5 * 32 (wmma k-steps)
#define NOUT    40    // 4 gates * 10
#define NPAD    48    // 3 * 16 (wmma n-tiles)
#define ROWS_PER_BLOCK 64
#define THREADS 128   // 4 wave32

union FragU { v16h h; uint4 q[2]; };

__global__ __launch_bounds__(THREADS) void qlstm_fused_kernel(
    const float* __restrict__ x,  const float* __restrict__ hx, const float* __restrict__ cx,
    const float* __restrict__ Wf, const float* __restrict__ bf,
    const float* __restrict__ Wi, const float* __restrict__ bi,
    const float* __restrict__ Wu, const float* __restrict__ bu,
    const float* __restrict__ Wo, const float* __restrict__ bo,
    float* __restrict__ out)
{
    __shared__ _Float16 Ash[ROWS_PER_BLOCK][KPAD];  // comb tile as f16, 20 KB
    __shared__ _Float16 Bsh[NPAD][KPAD];            // W^T (n-major) as f16, 15 KB
    __shared__ float    Pre[ROWS_PER_BLOCK][NPAD];  // pre-activations, 12 KB
    __shared__ float    Bias[NPAD];

    const int t = threadIdx.x;
    const int block_row0 = blockIdx.x * ROWS_PER_BLOCK;

    // ---- stage A: comb = [x | hx] padded with zeros to K=160, converted to f16 ----
    for (int idx = t; idx < ROWS_PER_BLOCK * KPAD; idx += THREADS) {
        const int r = idx / KPAD, k = idx % KPAD;
        const int grow = block_row0 + r;
        float v = 0.0f;
        if (k < IN_DIM)      v = x[grow * IN_DIM + k];
        else if (k < KDIM)   v = hx[grow * NQ + (k - IN_DIM)];
        Ash[r][k] = (_Float16)v;
    }
    // ---- stage B: Bsh[n][k] = W_gate[n%10][k]; rows 40..47 and k>=138 are zero ----
    for (int g = 0; g < 4; ++g) {
        const float* W = (g == 0) ? Wf : (g == 1) ? Wi : (g == 2) ? Wu : Wo;
        for (int idx = t; idx < NQ * KPAD; idx += THREADS) {
            const int j = idx / KPAD, k = idx % KPAD;
            Bsh[g * NQ + j][k] = (k < KDIM) ? (_Float16)W[j * KDIM + k] : (_Float16)0.0f;
        }
        const float* B = (g == 0) ? bf : (g == 1) ? bi : (g == 2) ? bu : bo;
        if (t < NQ) Bias[g * NQ + t] = B[t];
    }
    for (int idx = t; idx < (NPAD - NOUT) * KPAD; idx += THREADS)
        Bsh[NOUT + idx / KPAD][idx % KPAD] = (_Float16)0.0f;
    __syncthreads();

    // ---- WMMA: each wave computes a 16-row x 48-col tile of comb @ W^T ----
    const int wave  = t >> 5;
    const int lane  = t & 31;
    const int lidx  = lane & 15;              // M (for A) / N (for B) within tile
    const int khalf = (lane >> 4) ? 8 : 0;    // high-lane K offset per ISA layout

    v8f acc0 = {}, acc1 = {}, acc2 = {};
    const _Float16* arow = &Ash[wave * 16 + lidx][0];
    const _Float16* brow0 = &Bsh[lidx][0];
    const _Float16* brow1 = &Bsh[16 + lidx][0];
    const _Float16* brow2 = &Bsh[32 + lidx][0];
#pragma unroll
    for (int ks = 0; ks < KPAD / 32; ++ks) {
        const int kb = ks * 32 + khalf;
        FragU a, b0, b1, b2;
        a.q[0]  = *(const uint4*)(arow  + kb);  a.q[1]  = *(const uint4*)(arow  + kb + 16);
        b0.q[0] = *(const uint4*)(brow0 + kb);  b0.q[1] = *(const uint4*)(brow0 + kb + 16);
        b1.q[0] = *(const uint4*)(brow1 + kb);  b1.q[1] = *(const uint4*)(brow1 + kb + 16);
        b2.q[0] = *(const uint4*)(brow2 + kb);  b2.q[1] = *(const uint4*)(brow2 + kb + 16);
        acc0 = __builtin_amdgcn_wmma_f32_16x16x32_f16(false, a.h, false, b0.h, (short)0, acc0, false, false);
        acc1 = __builtin_amdgcn_wmma_f32_16x16x32_f16(false, a.h, false, b1.h, (short)0, acc1, false, false);
        acc2 = __builtin_amdgcn_wmma_f32_16x16x32_f16(false, a.h, false, b2.h, (short)0, acc2, false, false);
    }

    // ---- scatter D (VGPR r, lane: M = r + 8*(lane>=16), N = lane&15) to LDS ----
    {
        const int n0 = lane & 15;
        const int mb = (lane >> 4) * 8;
#pragma unroll
        for (int r = 0; r < 8; ++r) {
            const int m = wave * 16 + mb + r;
            Pre[m][n0]      = acc0[r];
            Pre[m][16 + n0] = acc1[r];
            Pre[m][32 + n0] = acc2[r];
        }
    }
    __syncthreads();

    // ---- epilogue: analytic qgate + LSTM cell, one lane per batch row ----
    // E_j = prod_{m<=j, m==j mod 2} cos(theta_m)  (RZ phases & CNOT permutations
    // cancel in the Z-expectations of the product state; closed form).
    if (t < ROWS_PER_BLOCK) {
        const int grow = block_row0 + t;
        float E[4][NQ];
#pragma unroll
        for (int g = 0; g < 4; ++g) {
#pragma unroll
            for (int j = 0; j < NQ; ++j) {
                const float th = Pre[t][g * NQ + j] + Bias[g * NQ + j];
                const float c  = cosf(th);
                E[g][j] = (j >= 2 ? E[g][j - 2] : 1.0f) * c;
            }
        }
#pragma unroll
        for (int j = 0; j < NQ; ++j) {
            const float fj = 1.0f / (1.0f + expf(-E[0][j]));
            const float ij = 1.0f / (1.0f + expf(-E[1][j]));
            const float gj = tanhf(E[2][j]);
            const float oj = 1.0f / (1.0f + expf(-E[3][j]));
            const float cn = fj * cx[grow * NQ + j] + ij * gj;
            const float hn = oj * tanhf(cn);
            out[grow * NQ + j]              = hn;   // h_new
            out[BATCH * NQ + grow * NQ + j] = cn;   // c_new
        }
    }
}

extern "C" void kernel_launch(void* const* d_in, const int* in_sizes, int n_in,
                              void* d_out, int out_size, void* d_ws, size_t ws_size,
                              hipStream_t stream) {
    (void)in_sizes; (void)n_in; (void)out_size; (void)d_ws; (void)ws_size;
    const float* x  = (const float*)d_in[0];
    const float* hx = (const float*)d_in[1];
    const float* cx = (const float*)d_in[2];
    const float* Wf = (const float*)d_in[3];
    const float* bf = (const float*)d_in[4];
    const float* Wi = (const float*)d_in[5];
    const float* bi = (const float*)d_in[6];
    const float* Wu = (const float*)d_in[7];
    const float* bu = (const float*)d_in[8];
    const float* Wo = (const float*)d_in[9];
    const float* bo = (const float*)d_in[10];
    float* out = (float*)d_out;

    dim3 grid(BATCH / ROWS_PER_BLOCK);   // 128 blocks
    dim3 block(THREADS);                 // 4 wave32 per block
    qlstm_fused_kernel<<<grid, block, 0, stream>>>(
        x, hx, cx, Wf, bf, Wi, bi, Wu, bu, Wo, bo, out);
}
